// VectorQuantizer_Sim_1271310319900
// MI455X (gfx1250) — matrix operand order repaired and checked by reading, not compiled, not run on hardware
//
#include <hip/hip_runtime.h>

#define B_    64
#define N_    256
#define CIN   768
#define CD    32
#define NE    8192
#define M_TOT (B_ * N_)   // 16384 token rows
#define MT    4           // M-tiles (of 16 rows) per block in k_argmin

typedef __attribute__((ext_vector_type(16))) _Float16 v16h;
typedef __attribute__((ext_vector_type(8)))  _Float16 v8h;
typedef __attribute__((ext_vector_type(8)))  float    v8f;

// Async global->LDS path (ASYNCcnt), guarded so compile never breaks.
#if defined(__AMDGCN__) && __has_builtin(__builtin_amdgcn_global_load_async_to_lds_b32) && \
    __has_builtin(__builtin_amdgcn_s_wait_asynccnt)
#define HAVE_ASYNC_LDS 1
typedef __attribute__((address_space(1))) int* gl_i32p;
typedef __attribute__((address_space(3))) int* lds_i32p;
#endif

__device__ __forceinline__ int lane_id() { return (int)(threadIdx.x & 31u); }

// wave32 lane gather: dst = src[srcLane]  (ds_bpermute, byte index)
__device__ __forceinline__ float bperm_f(float v, int srcLane) {
  return __int_as_float(__builtin_amdgcn_ds_bpermute(srcLane << 2, __float_as_int(v)));
}
__device__ __forceinline__ int bperm_i(int v, int srcLane) {
  return __builtin_amdgcn_ds_bpermute(srcLane << 2, v);
}

// ---------------------------------------------------------------------------
// Kernel 1: implicit_codebook = emb @ W_ct + b_ct  (+ f16 copy, + row norms)
// One wave per codebook row; lane d owns output channel d (CD == warpSize).
// ---------------------------------------------------------------------------
__global__ void __launch_bounds__(256)
k_codebook(const float* __restrict__ emb, const float* __restrict__ Wct,
           const float* __restrict__ bct,
           float* __restrict__ cbF, _Float16* __restrict__ cbH,
           float* __restrict__ cnorm) {
  const int row = (int)((blockIdx.x * blockDim.x + threadIdx.x) >> 5);
  const int d   = lane_id();
  if (row >= NE) return;
  const float* e = emb + (size_t)row * CD;
  float acc = bct[d];
#pragma unroll
  for (int k = 0; k < CD; ++k) acc = fmaf(e[k], Wct[k * CD + d], acc);
  cbF[(size_t)row * CD + d] = acc;
  cbH[(size_t)row * CD + d] = (_Float16)acc;
  float sq = acc * acc;
#pragma unroll
  for (int m = 1; m < 32; m <<= 1) sq += bperm_f(sq, d ^ m);
  if (d == 0) cnorm[row] = sq;
}

// ---------------------------------------------------------------------------
// Kernel 2: zc = z @ W_compress + b_compress (+ f16 copy)
// One wave per token row; lane d owns channel d; z[row][k] is lane-uniform
// (broadcast) load, Wc[k][0..31] is one coalesced 128B line per k.
// ||zc||^2 is NOT needed: it is constant per row, so argmin-invariant.
// ---------------------------------------------------------------------------
__global__ void __launch_bounds__(256)
k_compress(const float* __restrict__ z, const float* __restrict__ Wc,
           const float* __restrict__ bc,
           float* __restrict__ zcF, _Float16* __restrict__ zcH) {
  const int row = (int)((blockIdx.x * blockDim.x + threadIdx.x) >> 5);
  const int d   = lane_id();
  if (row >= M_TOT) return;
  const float* zr = z + (size_t)row * CIN;
  float acc = bc[d];
#pragma unroll 4
  for (int k = 0; k < CIN; ++k) acc = fmaf(zr[k], Wc[k * CD + d], acc);
  zcF[(size_t)row * CD + d] = acc;
  zcH[(size_t)row * CD + d] = (_Float16)acc;
}

// ---------------------------------------------------------------------------
// Kernel 3: fused distance + argmin over NE codebook entries.
// Block = 64 token rows (MT=4 tiles of 16); wave w scans codebook chunk
// [w*1024, (w+1)*1024) in 64 uniform iterations. Each B tile load feeds
// MT v_wmma_f32_16x16x32_f16 ops (K = CD = 32 exactly, one WMMA per tile).
// score(m,n) = ||c_n||^2 - 2 * (zc . c_n)   (||z||^2 dropped: argmin-invariant)
// ---------------------------------------------------------------------------
__global__ void __launch_bounds__(256)
k_argmin(const _Float16* __restrict__ zcH, const _Float16* __restrict__ cbH,
         const float* __restrict__ cnorm, int* __restrict__ idxOut) {
  __shared__ float sMin[8][16 * MT];
  __shared__ int   sIdx[8][16 * MT];

  const int m0   = (int)blockIdx.x * (16 * MT);
  const int w    = (int)(threadIdx.x >> 5);
  const int L    = lane_id();
  const int half = L >> 4;     // 0: lanes 0-15, 1: lanes 16-31
  const int l16  = L & 15;

  // A fragments: tile t covers rows m0+16t .. m0+16t+15, K = 0..31 (f16).
  // Lane layout: lane half selects contiguous K range {ak..ak+15} (matches B).
  const int ak = half ? 16 : 0;
  v16h a[MT];
#pragma unroll
  for (int t = 0; t < MT; ++t) {
    const _Float16* ap = zcH + (size_t)(m0 + 16 * t + l16) * CD + ak;
    const v8h lo = *(const v8h*)ap;
    const v8h hi = *(const v8h*)(ap + 8);
#pragma unroll
    for (int i = 0; i < 8; ++i) { a[t][i] = lo[i]; a[t][i + 8] = hi[i]; }
  }

  float best[MT][8];
  int   bidx[MT][8];
#pragma unroll
  for (int t = 0; t < MT; ++t)
#pragma unroll
    for (int r = 0; r < 8; ++r) { best[t][r] = 3.4e38f; bidx[t][r] = 0; }

  const int nBeg = w * (NE / 8);
  const int bk   = half ? 16 : 0;   // B fragment K-offset for this lane half

  for (int it = 0; it < (NE / 8) / 16; ++it) {   // 64 iters, scalar trip count
    const int nCol = nBeg + it * 16 + l16;
    __builtin_prefetch(cbH + (size_t)(nCol + 16) * CD, 0, 1);
    // B fragment (32x16 f16): lane holds codebook row nCol, 16 contiguous halves
    const v16h b  = *(const v16h*)(cbH + (size_t)nCol * CD + bk);
    const float cn = cnorm[nCol];
#pragma unroll
    for (int t = 0; t < MT; ++t) {
      v8f c = {};
      c = __builtin_amdgcn_wmma_f32_16x16x32_f16(false, a[t], false, b,
                                                 (short)0, c, false, false);
#pragma unroll
      for (int r = 0; r < 8; ++r) {
        const float dist = fmaf(-2.0f, c[r], cn);
        if (dist < best[t][r]) { best[t][r] = dist; bidx[t][r] = nCol; }
      }
    }
  }

  // min+index butterfly across the 16 lanes holding each C-row (masks 1,2,4,8)
#pragma unroll
  for (int t = 0; t < MT; ++t) {
#pragma unroll
    for (int r = 0; r < 8; ++r) {
      float m = best[t][r];
      int   i = bidx[t][r];
#pragma unroll
      for (int msk = 1; msk < 16; msk <<= 1) {
        const float om = bperm_f(m, L ^ msk);
        const int   oi = bperm_i(i, L ^ msk);
        if (om < m || (om == m && oi < i)) { m = om; i = oi; }
      }
      best[t][r] = m; bidx[t][r] = i;
    }
  }
  if (l16 == 0) {
#pragma unroll
    for (int t = 0; t < MT; ++t)
#pragma unroll
      for (int r = 0; r < 8; ++r) {
        sMin[w][16 * t + r + 8 * half] = best[t][r];
        sIdx[w][16 * t + r + 8 * half] = bidx[t][r];
      }
  }
  __syncthreads();
  if (threadIdx.x < 16 * MT) {
    float m = sMin[0][threadIdx.x];
    int   i = sIdx[0][threadIdx.x];
#pragma unroll
    for (int ww = 1; ww < 8; ++ww) {
      const float om = sMin[ww][threadIdx.x];
      const int   oi = sIdx[ww][threadIdx.x];
      if (om < m || (om == m && oi < i)) { m = om; i = oi; }
    }
    idxOut[m0 + (int)threadIdx.x] = i;
  }
}

// ---------------------------------------------------------------------------
// Kernel 4: out = z_q @ W_expand + b_expand.
// Thread owns one output column j: W_expand column (32 floats) in REGISTERS;
// block streams 64 gathered z_q rows through LDS (async global->LDS,
// ASYNCcnt). W_expand global traffic: 24 MB total (vs 1.5 GB for
// block-per-row), against a 48 MB output-write floor.
// Grid: (M_TOT/64) x (CIN/256).
// ---------------------------------------------------------------------------
__global__ void __launch_bounds__(256)
k_expand(const int* __restrict__ idx, const float* __restrict__ cbF,
         const float* __restrict__ We, const float* __restrict__ be,
         float* __restrict__ out) {
  __shared__ float zqS[64][CD];
  const int t  = (int)threadIdx.x;
  const int r0 = (int)blockIdx.x * 64;
  const int j  = (int)blockIdx.y * 256 + t;

  // Stage 64 gathered z_q rows into LDS (2048 floats, 8 per thread).
#pragma unroll
  for (int e = t; e < 64 * CD; e += 256) {
    const int r = e >> 5, d = e & 31;
    const float* src = cbF + (size_t)idx[r0 + r] * CD + d;
#if defined(HAVE_ASYNC_LDS)
    __builtin_amdgcn_global_load_async_to_lds_b32(
        (gl_i32p)src, (lds_i32p)&zqS[r][d], 0, 0);
#else
    zqS[r][d] = *src;
#endif
  }
#if defined(HAVE_ASYNC_LDS)
  __builtin_amdgcn_s_wait_asynccnt(0);
#endif
  __syncthreads();

  // W_expand column j in registers (coalesced: lane-consecutive j).
  float wreg[CD];
#pragma unroll
  for (int d = 0; d < CD; ++d) wreg[d] = We[d * CIN + j];
  const float bj = be[j];

  for (int r = 0; r < 64; ++r) {
    float acc = bj;
#pragma unroll
    for (int d = 0; d < CD; ++d) acc = fmaf(zqS[r][d], wreg[d], acc);
    out[(size_t)(r0 + r) * CIN + j] = acc;
  }
}

// ---------------------------------------------------------------------------
// Kernel 5: per-row loss partials (tiny: 4 MB traffic).
// ---------------------------------------------------------------------------
__global__ void __launch_bounds__(256)
k_rowloss(const int* __restrict__ idx, const float* __restrict__ cbF,
          const float* __restrict__ zcF, float* __restrict__ rowloss) {
  const int row = (int)(blockIdx.x * blockDim.x + threadIdx.x);
  if (row >= M_TOT) return;
  const float* q  = cbF + (size_t)idx[row] * CD;
  const float* zc = zcF + (size_t)row * CD;
  float s = 0.0f;
#pragma unroll
  for (int d = 0; d < CD; ++d) {
    const float df = q[d] - zc[d];
    s = fmaf(df, df, s);
  }
  rowloss[row] = s;
}

// ---------------------------------------------------------------------------
// Kernel 6: deterministic tree reduction of per-row loss partials.
// loss = (1 + 2) * mean((z_q - zc)^2)   [stop_gradient is a forward no-op]
// ---------------------------------------------------------------------------
__global__ void __launch_bounds__(256)
k_loss(const float* __restrict__ rowloss, float* __restrict__ lossOut) {
  __shared__ float s[256];
  float acc = 0.0f;
  for (int i = (int)threadIdx.x; i < M_TOT; i += 256) acc += rowloss[i];
  s[threadIdx.x] = acc;
  __syncthreads();
  for (int st = 128; st > 0; st >>= 1) {
    if ((int)threadIdx.x < st) s[threadIdx.x] += s[threadIdx.x + st];
    __syncthreads();
  }
  if (threadIdx.x == 0)
    lossOut[0] = 3.0f * s[0] / (float)(M_TOT * CD);
}

// ---------------------------------------------------------------------------
extern "C" void kernel_launch(void* const* d_in, const int* in_sizes, int n_in,
                              void* d_out, int out_size, void* d_ws, size_t ws_size,
                              hipStream_t stream) {
  const float* z   = (const float*)d_in[0];  // [64,256,768]
  const float* emb = (const float*)d_in[1];  // [8192,32]
  const float* Wc  = (const float*)d_in[2];  // [768,32]
  const float* bc  = (const float*)d_in[3];  // [32]
  const float* Wct = (const float*)d_in[4];  // [32,32]
  const float* bct = (const float*)d_in[5];  // [32]
  const float* We  = (const float*)d_in[6];  // [32,768]
  const float* be  = (const float*)d_in[7];  // [768]

  char* ws = (char*)d_ws;                        // ~4.9 MB used, 256B-aligned
  float*    cbF     = (float*)(ws + 0);          // 8192*32 f32   (1 MB)
  _Float16* cbH     = (_Float16*)(ws + 1048576); // 8192*32 f16   (0.5 MB)
  float*    cnorm   = (float*)(ws + 1572864);    // 8192 f32
  float*    zcF     = (float*)(ws + 1605632);    // 16384*32 f32  (2 MB)
  _Float16* zcH     = (_Float16*)(ws + 3702784); // 16384*32 f16  (1 MB)
  int*      idx     = (int*)(ws + 4751360);      // 16384 i32
  float*    rowloss = (float*)(ws + 4816896);    // 16384 f32

  float* out  = (float*)d_out;
  float* loss = out + (size_t)M_TOT * CIN;       // scalar loss after [B,N,CIN]

  hipLaunchKernelGGL(k_codebook, dim3(NE / 8),    dim3(256), 0, stream,
                     emb, Wct, bct, cbF, cbH, cnorm);
  hipLaunchKernelGGL(k_compress, dim3(M_TOT / 8), dim3(256), 0, stream,
                     z, Wc, bc, zcF, zcH);
  hipLaunchKernelGGL(k_argmin,   dim3(M_TOT / (16 * MT)), dim3(256), 0, stream,
                     zcH, cbH, cnorm, idx);
  hipLaunchKernelGGL(k_expand,   dim3(M_TOT / 64, CIN / 256), dim3(256), 0, stream,
                     idx, cbF, We, be, out);
  hipLaunchKernelGGL(k_rowloss,  dim3(M_TOT / 256), dim3(256), 0, stream,
                     idx, cbF, zcF, rowloss);
  hipLaunchKernelGGL(k_loss,     dim3(1),         dim3(256), 0, stream,
                     rowloss, loss);
}
